// MultiHeadAttention_7980049236244
// MI455X (gfx1250) — matrix operand order repaired and checked by reading, not compiled, not run on hardware
//
#include <hip/hip_runtime.h>

// =====================================================================
// Fused MHA for MI455X (gfx1250, wave32, WMMA).
//   f32->bf16 convert -> QKV GEMM (double-buffered LDS, wmma bf16, epilogue
//   scatters Q/K packed + V transposed) -> flash attention (wmma bf16,
//   online softmax: rowmax via v_permlane16, rowsum via ones-WMMA)
//   -> output projection GEMM (wmma, f32 out).
// Workspace (bf16 halves): xb[4096*1024] | wqkvb[3072*1024] | wob[1024*1024]
//   | qk[4096*2048] | vt[1024*4096] | vals[4096*1024]   (~50.3 MB)
// =====================================================================

typedef __attribute__((ext_vector_type(16))) __bf16        v16bf;
typedef __attribute__((ext_vector_type(8)))  float         v8f;
typedef __attribute__((ext_vector_type(4)))  unsigned int  u32x4;   // POD 128-bit

union FragAB { v16bf v; unsigned int u[8]; u32x4 q[2]; unsigned short s[16]; };
union FragC  { v8f   v; float f[8]; };

__device__ __forceinline__ unsigned short f2bf(float x) {
  union { float f; unsigned int u; } c; c.f = x;
  unsigned int u = c.u;
  u += 0x7FFFu + ((u >> 16) & 1u);   // round-to-nearest-even
  return (unsigned short)(u >> 16);
}

__device__ __forceinline__ v8f wmma_bf16(const FragAB& a, const FragAB& b, v8f c) {
  return __builtin_amdgcn_wmma_f32_16x16x32_bf16(false, a.v, false, b.v,
                                                 (short)0, c, false, false);
}

// xor-butterfly max across 16-lane groups.  VALU (v_permlane16_b32) when
// available; ds_bpermute-based __shfl_xor fallback otherwise.
#if __has_builtin(__builtin_amdgcn_permlane16)
__device__ __forceinline__ float pl16(float v, unsigned sel0, unsigned sel1) {
  union { float f; unsigned u; } a, r;
  a.f = v;
  r.u = __builtin_amdgcn_permlane16(a.u, a.u, sel0, sel1, false, false);
  return r.f;
}
__device__ __forceinline__ float rowmax16(float v) {
  v = fmaxf(v, pl16(v, 0x67452301u, 0xEFCDAB89u));  // xor 1
  v = fmaxf(v, pl16(v, 0x54761032u, 0xDCFE98BAu));  // xor 2
  v = fmaxf(v, pl16(v, 0x32107654u, 0xBA98FEDCu));  // xor 4
  v = fmaxf(v, pl16(v, 0xFEDCBA98u, 0x76543210u));  // xor 8
  return v;
}
#else
__device__ __forceinline__ float rowmax16(float v) {
#pragma unroll
  for (int d = 1; d < 16; d <<= 1) v = fmaxf(v, __shfl_xor(v, d, 32));
  return v;
}
#endif

// ---------------------------------------------------------------------
__global__ __launch_bounds__(256)
void f32_to_bf16_k(const float* __restrict__ in, unsigned short* __restrict__ out, int n) {
  int i = blockIdx.x * blockDim.x + threadIdx.x;
  int stride = gridDim.x * blockDim.x;
  for (; i < n; i += stride) out[i] = f2bf(in[i]);
}

// ---------------------------------------------------------------------
// C(MxN) = A(MxK bf16 rm) * W(NxK bf16 rm)^T.  Block 64x64, BK=32,
// 128 threads / 4 waves, 2x2 16x16 WMMA tiles per wave.
// Double-buffered LDS, one barrier per K-step.
// MODE 0: QKV epilogue  (Q/K packed bf16, V transposed bf16, b128 stores)
// MODE 1: plain f32 out
// ---------------------------------------------------------------------
template <int MODE>
__global__ __launch_bounds__(128)
void gemm_bf16_nt(const unsigned short* __restrict__ A,
                  const unsigned short* __restrict__ W,
                  unsigned short* __restrict__ qk,
                  unsigned short* __restrict__ vt,
                  float* __restrict__ cf,
                  int M, int N, int K) {
  __shared__ __align__(16) unsigned short sA[2][64 * 40];  // 80B rows, 16B aligned
  __shared__ __align__(16) unsigned short sW[2][64 * 40];

  const int tid  = threadIdx.x;
  const int wave = tid >> 5;
  const int lane = tid & 31;
  const int half = lane >> 4;
  const int ln   = lane & 15;

  const int m0 = blockIdx.y * 64;
  const int n0 = blockIdx.x * 64;
  const int wm = (wave >> 1) * 32;
  const int wn = (wave & 1) * 32;

  // staging: 2 threads per row, 2x b128 each
  const int srow = tid >> 1;
  const int sseg = (tid & 1) * 16;
  const unsigned short* agbase = A + (size_t)(m0 + srow) * K + sseg;
  const unsigned short* wgbase = W + (size_t)(n0 + srow) * K + sseg;

  FragC acc[2][2];
#pragma unroll
  for (int i = 0; i < 2; ++i)
#pragma unroll
    for (int j = 0; j < 2; ++j)
#pragma unroll
      for (int r = 0; r < 8; ++r) acc[i][j].f[r] = 0.0f;

  // prologue: stage tile k0=0 into buffer 0
  u32x4 a0 = *(const u32x4*)(agbase);
  u32x4 a1 = *(const u32x4*)(agbase + 8);
  u32x4 w0 = *(const u32x4*)(wgbase);
  u32x4 w1 = *(const u32x4*)(wgbase + 8);
  *(u32x4*)(&sA[0][srow * 40 + sseg])     = a0;
  *(u32x4*)(&sA[0][srow * 40 + sseg + 8]) = a1;
  *(u32x4*)(&sW[0][srow * 40 + sseg])     = w0;
  *(u32x4*)(&sW[0][srow * 40 + sseg + 8]) = w1;
  __syncthreads();

  int p = 0;
  for (int k0 = 0; k0 < K; k0 += 32) {
    const bool more = (k0 + 32) < K;
    if (more) {   // fetch next tile into registers (overlaps WMMA below)
      a0 = *(const u32x4*)(agbase + k0 + 32);
      a1 = *(const u32x4*)(agbase + k0 + 40);
      w0 = *(const u32x4*)(wgbase + k0 + 32);
      w1 = *(const u32x4*)(wgbase + k0 + 40);
    }

    FragAB af[2], bw[2];
#pragma unroll
    for (int i = 0; i < 2; ++i) {           // A-frag: lane = row M
      int m = wm + i * 16 + ln;
      af[i].q[0] = *(const u32x4*)(&sA[p][m * 40 + half * 8]);
      af[i].q[1] = *(const u32x4*)(&sA[p][m * 40 + 16 + half * 8]);
    }
#pragma unroll
    for (int j = 0; j < 2; ++j) {           // B-frag: lane = col N, K table order
      int n = wn + j * 16 + ln;
      bw[j].q[0] = *(const u32x4*)(&sW[p][n * 40 + half * 8]);
      bw[j].q[1] = *(const u32x4*)(&sW[p][n * 40 + 16 + half * 8]);
    }
#pragma unroll
    for (int i = 0; i < 2; ++i)
#pragma unroll
      for (int j = 0; j < 2; ++j)
        acc[i][j].v = wmma_bf16(af[i], bw[j], acc[i][j].v);

    if (more) {   // stage next tile into the other buffer
      *(u32x4*)(&sA[p ^ 1][srow * 40 + sseg])     = a0;
      *(u32x4*)(&sA[p ^ 1][srow * 40 + sseg + 8]) = a1;
      *(u32x4*)(&sW[p ^ 1][srow * 40 + sseg])     = w0;
      *(u32x4*)(&sW[p ^ 1][srow * 40 + sseg + 8]) = w1;
    }
    __syncthreads();
    p ^= 1;
  }

  // D layout: lane -> N = ln, VGPR r -> M = r + 8*half
#pragma unroll
  for (int i = 0; i < 2; ++i)
#pragma unroll
    for (int j = 0; j < 2; ++j) {
      const int fbase = n0 + wn + j * 16;            // uniform per tile
      if (MODE == 1) {
#pragma unroll
        for (int r = 0; r < 8; ++r) {
          int mg = m0 + wm + i * 16 + r + 8 * half;
          cf[(size_t)mg * N + fbase + ln] = acc[i][j].f[r];
        }
      } else {
        const int h    = fbase / 192;
        const int fmod = fbase % 192;                // 16-aligned, single region
        if (fmod < 128) {
          const int col = h * 128 + fmod + ln;       // Q: 0..63, K: 64..127
#pragma unroll
          for (int r = 0; r < 8; ++r) {
            int mg = m0 + wm + i * 16 + r + 8 * half;
            qk[(size_t)mg * 2048 + col] = f2bf(acc[i][j].f[r]);
          }
        } else {
          const int drow  = h * 64 + (fmod - 128) + ln;
          const int mbase = m0 + wm + i * 16 + 8 * half;  // 8 consecutive m
          u32x4 pk;
#pragma unroll
          for (int t = 0; t < 4; ++t)
            pk[t] = (unsigned int)f2bf(acc[i][j].f[2 * t]) |
                    ((unsigned int)f2bf(acc[i][j].f[2 * t + 1]) << 16);
          *(u32x4*)(vt + (size_t)drow * 4096 + mbase) = pk;
        }
      }
    }
}

// ---------------------------------------------------------------------
// Flash attention. Q/K from qk[(m, h*128+{0,64}+d)], V from vt[(h*64+d, m)].
// Grid (S/64, H, B); 128 threads = 4 waves; wave owns 16 query rows.
// Per 32-key chunk: 4 wmma logits + rowmax (permlane16) + exp +
//                   1 wmma rowsum + 4 wmma P*V.
// ---------------------------------------------------------------------
__global__ __launch_bounds__(128)
void attn_flash(const unsigned short* __restrict__ qk,
                const unsigned short* __restrict__ vt,
                const unsigned char* __restrict__ mask,   // (B*S) bool, true=keep
                unsigned short* __restrict__ vals) {      // (B*S) x 1024 bf16
  constexpr int S  = 2048;
  constexpr int QS = 2048;   // qk row stride
  constexpr int E  = 1024;
  const float scale = 0.125f;  // 1/sqrt(64)

  const int b    = blockIdx.z;
  const int h    = blockIdx.y;
  const int qblk = blockIdx.x;

  const int tid  = threadIdx.x;
  const int wave = tid >> 5;
  const int lane = tid & 31;
  const int half = lane >> 4;
  const int ln   = lane & 15;

  const int qrow = qblk * 64 + wave * 16;

  // per-wave P staging (C-layout -> A-layout); in-wave DS ops are in-order
  // per the ISA so no workgroup barrier is needed.
  __shared__ __align__(16) unsigned short sP[4][16 * 40];
  unsigned short* pw = &sP[wave][0];

  // Q: two 16x32 A-fragments, two b128 loads each
  FragAB qf[2];
  {
    const unsigned short* qp = qk + (size_t)(b * S + qrow + ln) * QS + h * 128;
#pragma unroll
    for (int c = 0; c < 2; ++c) {
      qf[c].q[0] = *(const u32x4*)(qp + c * 32 + half * 8);
      qf[c].q[1] = *(const u32x4*)(qp + c * 32 + 16 + half * 8);
    }
  }

  // constant all-ones bf16 B-fragment for WMMA row-sum
  FragAB onesb;
#pragma unroll
  for (int v = 0; v < 8; ++v) onesb.u[v] = 0x3F803F80u;

  FragC o[4];
  float rm[8], rl[8];
#pragma unroll
  for (int c = 0; c < 4; ++c)
#pragma unroll
    for (int r = 0; r < 8; ++r) o[c].f[r] = 0.0f;
#pragma unroll
  for (int r = 0; r < 8; ++r) { rm[r] = -3.0e38f; rl[r] = 0.0f; }

  for (int kk0 = 0; kk0 < S; kk0 += 32) {
    // ---- logits over keys [kk0, kk0+32) ----
    FragC s0, s1;
#pragma unroll
    for (int r = 0; r < 8; ++r) { s0.f[r] = 0.0f; s1.f[r] = 0.0f; }
#pragma unroll
    for (int c = 0; c < 2; ++c) {
      FragAB kb0, kb1;   // B-frag: lane = key, K dim = d (table order)
      const unsigned short* kp0 =
          qk + (size_t)(b * S + kk0 + ln) * QS + h * 128 + 64 + c * 32 + half * 8;
      const unsigned short* kp1 = kp0 + (size_t)16 * QS;
      kb0.q[0] = *(const u32x4*)(kp0);
      kb0.q[1] = *(const u32x4*)(kp0 + 16);
      kb1.q[0] = *(const u32x4*)(kp1);
      kb1.q[1] = *(const u32x4*)(kp1 + 16);
      s0.v = wmma_bf16(qf[c], kb0, s0.v);
      s1.v = wmma_bf16(qf[c], kb1, s1.v);
    }

    // prefetch next chunk's K rows and Vt rows (global_prefetch_b8)
    if (kk0 + 32 < S) {
      __builtin_prefetch(qk + (size_t)(b * S + kk0 + 32 + lane) * QS + h * 128 + 64, 0, 1);
      const unsigned short* vpf =
          vt + (size_t)(h * 64 + 2 * lane) * 4096 + (size_t)b * S + kk0 + 32;
      __builtin_prefetch(vpf, 0, 1);
      __builtin_prefetch(vpf + 4096, 0, 1);
    }

    const bool keep0 = mask[b * S + kk0 + ln] != 0;
    const bool keep1 = mask[b * S + kk0 + 16 + ln] != 0;

    // ---- online softmax (row = r + 8*half, spread over a 16-lane half) ----
#pragma unroll
    for (int r = 0; r < 8; ++r) {
      float a0 = keep0 ? s0.f[r] * scale : -3.0e38f;
      float a1 = keep1 ? s1.f[r] * scale : -3.0e38f;
      float mx = rowmax16(fmaxf(a0, a1));
      float mnew  = fmaxf(rm[r], mx);
      float alpha = __expf(rm[r] - mnew);
      s0.f[r] = keep0 ? __expf(a0 - mnew) : 0.0f;
      s1.f[r] = keep1 ? __expf(a1 - mnew) : 0.0f;
      rm[r] = mnew;
      rl[r] *= alpha;                      // row-sum added below via WMMA
#pragma unroll
      for (int c = 0; c < 4; ++c) o[c].f[r] *= alpha;
    }

    // ---- P (16x32, C-layout) -> LDS -> bf16 A-fragment ----
#pragma unroll
    for (int r = 0; r < 8; ++r) {
      int m = r + 8 * half;
      pw[m * 40 + ln]      = f2bf(s0.f[r]);
      pw[m * 40 + 16 + ln] = f2bf(s1.f[r]);
    }
    FragAB pf;
    pf.q[0] = *(const u32x4*)(&pw[ln * 40 + half * 8]);
    pf.q[1] = *(const u32x4*)(&pw[ln * 40 + 16 + half * 8]);

    // ---- row-sum of P via WMMA with all-ones B ----
    FragC ps;
#pragma unroll
    for (int r = 0; r < 8; ++r) ps.f[r] = 0.0f;
    ps.v = wmma_bf16(pf, onesb, ps.v);
#pragma unroll
    for (int r = 0; r < 8; ++r) rl[r] += ps.f[r];

    // ---- O += P * V ; vt rows key-major -> contiguous b128 loads ----
#pragma unroll
    for (int c = 0; c < 4; ++c) {
      FragAB vb;   // B-frag: lane = d col, K dim = key (table order)
      const unsigned short* vp =
          vt + (size_t)(h * 64 + c * 16 + ln) * 4096 + (size_t)b * S + kk0 + half * 8;
      vb.q[0] = *(const u32x4*)(vp);
      vb.q[1] = *(const u32x4*)(vp + 16);
      o[c].v = wmma_bf16(pf, vb, o[c].v);
    }
  }

  // ---- normalize, store values (B,S,E) bf16 for the out-proj GEMM ----
#pragma unroll
  for (int c = 0; c < 4; ++c)
#pragma unroll
    for (int r = 0; r < 8; ++r) {
      int m = qrow + r + 8 * half;
      float val = o[c].f[r] / rl[r];
      vals[(size_t)(b * S + m) * E + h * 64 + c * 16 + ln] = f2bf(val);
    }
}

// ---------------------------------------------------------------------
extern "C" void kernel_launch(void* const* d_in, const int* in_sizes, int n_in,
                              void* d_out, int out_size, void* d_ws, size_t ws_size,
                              hipStream_t stream) {
  (void)in_sizes; (void)n_in; (void)out_size; (void)ws_size;

  const float*         x     = (const float*)d_in[0];
  const unsigned char* mask  = (const unsigned char*)d_in[1];  // jax bool = 1 byte
  const float*         w_qkv = (const float*)d_in[2];
  const float*         w_o   = (const float*)d_in[3];
  float*               out   = (float*)d_out;

  const int M  = 4096;   // B*S
  const int E  = 1024;
  const int Fq = 3072;   // 3*E

  unsigned short* xb    = (unsigned short*)d_ws;
  unsigned short* wqkvb = xb    + (size_t)M  * E;       // 3072x1024
  unsigned short* wob   = wqkvb + (size_t)Fq * E;       // 1024x1024
  unsigned short* qkbuf = wob   + (size_t)E  * E;       // 4096x2048 (Q|K per head)
  unsigned short* vtbuf = qkbuf + (size_t)M  * 2048;    // 1024x4096 (V transposed)
  unsigned short* vals  = vtbuf + (size_t)1024 * M;     // 4096x1024

  f32_to_bf16_k<<<2048, 256, 0, stream>>>(x,     xb,    M * E);
  f32_to_bf16_k<<<2048, 256, 0, stream>>>(w_qkv, wqkvb, Fq * E);
  f32_to_bf16_k<<<1024, 256, 0, stream>>>(w_o,   wob,   E * E);

  // QKV projection with scatter epilogue (Q/K packed, V transposed)
  gemm_bf16_nt<0><<<dim3(Fq / 64, M / 64), 128, 0, stream>>>(
      xb, wqkvb, qkbuf, vtbuf, nullptr, M, Fq, E);

  // attention: grid (S/64, H, B)
  attn_flash<<<dim3(32, 16, 2), 128, 0, stream>>>(qkbuf, vtbuf, mask, vals);

  // output projection -> f32
  gemm_bf16_nt<1><<<dim3(E / 64, M / 64), 128, 0, stream>>>(
      vals, wob, nullptr, nullptr, out, M, E, E);
}
